// SeqSelfAttention_19086834663496
// MI455X (gfx1250) — matrix operand order repaired and checked by reading, not compiled, not run on hardware
//
#include <hip/hip_runtime.h>
#include <hip/hip_bf16.h>

#define BB 2
#define LL 1024
#define DD 256
#define UU 32

typedef __attribute__((ext_vector_type(16))) __bf16 v16bf;
typedef __attribute__((ext_vector_type(8)))  float  v8f;

union FragU {
    v16bf v;
    uint4 u[2];
};

__device__ __forceinline__ float tanh_fast(float x) {
    // tanh(x) = 1 - 2/(exp(2x)+1); saturates correctly at +-1 for large |x|
    float e = __expf(2.0f * x);
    return 1.0f - 2.0f / (e + 1.0f);
}

// ---------------------------------------------------------------------------
// K1: q[b,l,u] = sum_d X[b,l,d]*Wt[d,u];  k[b,l,u] = sum_d X[b,l,d]*Wx[u,d]
// One 64-thread block (2 waves) per (b,l) row; row staged in LDS.
// ---------------------------------------------------------------------------
__global__ void __launch_bounds__(64) proj_qk_kernel(
    const float* __restrict__ X, const float* __restrict__ Wt,
    const float* __restrict__ Wx, float* __restrict__ q, float* __restrict__ k)
{
    __shared__ float row[DD];
    const int bl = blockIdx.x;                 // 0 .. B*L-1
    for (int i = threadIdx.x; i < DD; i += 64)
        row[i] = X[(size_t)bl * DD + i];
    __syncthreads();

    const int u = threadIdx.x;
    if (u < UU) {
        float acc = 0.0f;
        #pragma unroll 8
        for (int d = 0; d < DD; ++d)
            acc = fmaf(row[d], Wt[d * UU + u], acc);
        q[(size_t)bl * UU + u] = acc;
    } else {
        const int uu = u - UU;
        const float* w = Wx + (size_t)uu * DD;
        float acc = 0.0f;
        #pragma unroll 8
        for (int d = 0; d < DD; ++d)
            acc = fmaf(row[d], w[d], acc);
        k[(size_t)bl * UU + uu] = acc;
    }
}

// ---------------------------------------------------------------------------
// K2: Vt[b][d][l] = (bf16) X[b][l][d]   (column-major values for WMMA B-frags)
// ---------------------------------------------------------------------------
__global__ void __launch_bounds__(256) transpose_vals_kernel(
    const float* __restrict__ X, __hip_bfloat16* __restrict__ Vt)
{
    const size_t idx = (size_t)blockIdx.x * 256 + threadIdx.x;  // B*L*D threads
    const int d = (int)(idx % DD);
    const int l = (int)((idx / DD) % LL);
    const int b = (int)(idx / ((size_t)DD * LL));
    Vt[((size_t)b * DD + d) * LL + l] = __float2bfloat16(X[idx]);
}

// ---------------------------------------------------------------------------
// K3: fused alpha + stable softmax for one (b, q-row).
//   alpha[ki] = ba + sum_u Wa[u]*tanh(qv[u]+bh[u] + k[ki,u])
//   a = softmax(alpha) -> bf16 probs (row-major, K contiguous = WMMA A layout)
// 256 threads (8 wave32), 4 keys per thread.
// ---------------------------------------------------------------------------
__global__ void __launch_bounds__(256) alpha_softmax_kernel(
    const float* __restrict__ q, const float* __restrict__ kk,
    const float* __restrict__ bh, const float* __restrict__ Wa,
    const float* __restrict__ ba, __hip_bfloat16* __restrict__ aout)
{
    __shared__ float qv[UU];
    __shared__ float wa[UU];
    __shared__ float red[8];

    const int qi  = blockIdx.x;
    const int b   = blockIdx.y;
    const int tid = threadIdx.x;

    if (tid < UU) {
        qv[tid] = q[((size_t)b * LL + qi) * UU + tid] + bh[tid];
        wa[tid] = Wa[tid];
    }
    __syncthreads();

    const float* krow = kk + (size_t)b * LL * UU;
    float av[4];
    #pragma unroll
    for (int j = 0; j < 4; ++j) {
        const int ki = j * 256 + tid;
        const float4* kv4 = (const float4*)(krow + (size_t)ki * UU);
        float acc = 0.0f;
        #pragma unroll
        for (int u4 = 0; u4 < UU / 4; ++u4) {
            const float4 kv = kv4[u4];
            const int ub = u4 * 4;
            acc = fmaf(wa[ub + 0], tanh_fast(qv[ub + 0] + kv.x), acc);
            acc = fmaf(wa[ub + 1], tanh_fast(qv[ub + 1] + kv.y), acc);
            acc = fmaf(wa[ub + 2], tanh_fast(qv[ub + 2] + kv.z), acc);
            acc = fmaf(wa[ub + 3], tanh_fast(qv[ub + 3] + kv.w), acc);
        }
        av[j] = acc + ba[0];
    }

    // ---- block max (wave32 shuffle tree + LDS across 8 waves) ----
    const int wave = tid >> 5, lane = tid & 31;
    float m = fmaxf(fmaxf(av[0], av[1]), fmaxf(av[2], av[3]));
    #pragma unroll
    for (int off = 16; off > 0; off >>= 1)
        m = fmaxf(m, __shfl_xor(m, off, 32));
    if (lane == 0) red[wave] = m;
    __syncthreads();
    if (tid == 0) {
        float mm = red[0];
        #pragma unroll
        for (int i = 1; i < 8; ++i) mm = fmaxf(mm, red[i]);
        red[0] = mm;
    }
    __syncthreads();
    m = red[0];
    __syncthreads();   // everyone has read red[0] before it is reused

    // ---- exp + block sum ----
    float s = 0.0f;
    #pragma unroll
    for (int j = 0; j < 4; ++j) {
        av[j] = __expf(av[j] - m);
        s += av[j];
    }
    #pragma unroll
    for (int off = 16; off > 0; off >>= 1)
        s += __shfl_xor(s, off, 32);
    if (lane == 0) red[wave] = s;
    __syncthreads();
    if (tid == 0) {
        float ss = red[0];
        #pragma unroll
        for (int i = 1; i < 8; ++i) ss += red[i];
        red[0] = ss;
    }
    __syncthreads();
    const float inv = 1.0f / red[0];

    __hip_bfloat16* arow = aout + ((size_t)b * LL + qi) * LL;
    #pragma unroll
    for (int j = 0; j < 4; ++j)
        arow[j * 256 + tid] = __float2bfloat16(av[j] * inv);
}

// ---------------------------------------------------------------------------
// K4: c[b,q,d] = sum_k a[b,q,k] * V[b,k,d]  via v_wmma_f32_16x16x32_bf16.
// Grid: (L/16, B). Block: 128 threads = 4 wave32.
// Each wave: one 16(M) x 64(N) strip = 4 16x16 accumulators, K walked in 32s.
// Fragment layouts used here (contiguous K order, same for A and B):
//   A 16x32 bf16: lane%16 = row M; lanes<16 hold K {k0..k0+15},
//                 lanes>=16 hold K {k0+16..k0+31}
//   B 32x16 bf16: lane%16 = col N; lanes<16 hold K {k0..k0+15},
//                 lanes>=16 hold K {k0+16..k0+31}
//   C/D f32 16x16: N = lane%16; VGPR r -> M = r + 8*(lane/16)
// ---------------------------------------------------------------------------
__global__ void __launch_bounds__(128) wmma_av_gemm_kernel(
    const __hip_bfloat16* __restrict__ A,   // [B][L][L] probs (bf16)
    const __hip_bfloat16* __restrict__ Vt,  // [B][D][L] values^T (bf16)
    float* __restrict__ out)                // [B][L][D]
{
    const int mt   = blockIdx.x;        // 0..63 : q-tile
    const int b    = blockIdx.y;
    const int tid  = threadIdx.x;
    const int wave = tid >> 5;          // 0..3 : N strip of 64
    const int lane = tid & 31;
    const int half = lane >> 4;         // 0/1
    const int l15  = lane & 15;

    const __hip_bfloat16* Arow = A + (size_t)b * LL * LL + ((size_t)mt * 16 + l15) * LL;
    const __hip_bfloat16* Vb   = Vt + (size_t)b * DD * LL;
    const int n0 = wave * 64;
    const __hip_bfloat16* B0 = Vb + (size_t)(n0 +  0 + l15) * LL;
    const __hip_bfloat16* B1 = Vb + (size_t)(n0 + 16 + l15) * LL;
    const __hip_bfloat16* B2 = Vb + (size_t)(n0 + 32 + l15) * LL;
    const __hip_bfloat16* B3 = Vb + (size_t)(n0 + 48 + l15) * LL;

    v8f acc0 = {}, acc1 = {}, acc2 = {}, acc3 = {};

    #pragma unroll 4
    for (int k0 = 0; k0 < LL; k0 += 32) {
        const int ka = k0 + half * 16;   // A-frag K base for this lane half
        const int kb = k0 + half * 16;   // B-frag K base for this lane half

        FragU af;
        af.u[0] = *(const uint4*)(Arow + ka);
        af.u[1] = *(const uint4*)(Arow + ka + 8);

        FragU bf0, bf1, bf2, bf3;
        bf0.u[0] = *(const uint4*)(B0 + kb);  bf0.u[1] = *(const uint4*)(B0 + kb + 8);
        bf1.u[0] = *(const uint4*)(B1 + kb);  bf1.u[1] = *(const uint4*)(B1 + kb + 8);
        bf2.u[0] = *(const uint4*)(B2 + kb);  bf2.u[1] = *(const uint4*)(B2 + kb + 8);
        bf3.u[0] = *(const uint4*)(B3 + kb);  bf3.u[1] = *(const uint4*)(B3 + kb + 8);

        acc0 = __builtin_amdgcn_wmma_f32_16x16x32_bf16(false, af.v, false, bf0.v,
                                                       (short)0, acc0, false, false);
        acc1 = __builtin_amdgcn_wmma_f32_16x16x32_bf16(false, af.v, false, bf1.v,
                                                       (short)0, acc1, false, false);
        acc2 = __builtin_amdgcn_wmma_f32_16x16x32_bf16(false, af.v, false, bf2.v,
                                                       (short)0, acc2, false, false);
        acc3 = __builtin_amdgcn_wmma_f32_16x16x32_bf16(false, af.v, false, bf3.v,
                                                       (short)0, acc3, false, false);
    }

    float* orow = out + (size_t)b * LL * DD;
    const int mbase = mt * 16 + half * 8;
    #pragma unroll
    for (int r = 0; r < 8; ++r) {
        float* o = orow + (size_t)(mbase + r) * DD + l15;
        o[n0 +  0] = acc0[r];
        o[n0 + 16] = acc1[r];
        o[n0 + 32] = acc2[r];
        o[n0 + 48] = acc3[r];
    }
}

// ---------------------------------------------------------------------------
extern "C" void kernel_launch(void* const* d_in, const int* in_sizes, int n_in,
                              void* d_out, int out_size, void* d_ws, size_t ws_size,
                              hipStream_t stream)
{
    const float* X  = (const float*)d_in[0];   // [B,L,D]
    const float* Wt = (const float*)d_in[1];   // [D,U]
    const float* Wx = (const float*)d_in[2];   // [U,D]
    const float* bh = (const float*)d_in[3];   // [U]
    const float* Wa = (const float*)d_in[4];   // [U,1]
    const float* ba = (const float*)d_in[5];   // [1]
    float* out = (float*)d_out;                // [B,L,D]

    // workspace: q | k | probs(bf16) | Vt(bf16)  (~5.5 MB)
    float* q  = (float*)d_ws;                            // B*L*U f32
    float* kk = q + (size_t)BB * LL * UU;                // B*L*U f32
    __hip_bfloat16* aP = (__hip_bfloat16*)(kk + (size_t)BB * LL * UU); // B*L*L bf16
    __hip_bfloat16* Vt = aP + (size_t)BB * LL * LL;      // B*D*L bf16

    proj_qk_kernel<<<BB * LL, 64, 0, stream>>>(X, Wt, Wx, q, kk);

    transpose_vals_kernel<<<(BB * LL * DD) / 256, 256, 0, stream>>>(X, Vt);

    dim3 gAlpha(LL, BB);
    alpha_softmax_kernel<<<gAlpha, 256, 0, stream>>>(q, kk, bh, Wa, ba, aP);

    dim3 gGemm(LL / 16, BB);
    wmma_av_gemm_kernel<<<gGemm, 128, 0, stream>>>(aP, Vt, out);
}
